// MultiHeadAttention_17265768530789
// MI455X (gfx1250) — matrix operand order repaired and checked by reading, not compiled, not run on hardware
//
#include <hip/hip_runtime.h>
#include <stdint.h>

// ---------------------------------------------------------------------------
// MI455X (gfx1250) multi-head attention forward.
// f16 WMMA (v_wmma_f32_16x16x32_f16) everywhere, fp32 accumulate.
// Double-buffered async global->LDS pipelines (global_load_async_to_lds_b128
// + s_wait_asynccnt), non-temporal streaming of the 268MB attn output.
// ---------------------------------------------------------------------------

typedef _Float16 h16;
typedef __attribute__((ext_vector_type(16))) _Float16 v16h;
typedef __attribute__((ext_vector_type(8)))  float    v8f;

union F16x16 { v16h h; uint4 u[2]; _Float16 e[16]; };
union F32x8  { v8f  v; float f[8]; };

#define NHEAD 8
#define DMODEL 512
#define DK 64
#define BATCH 2
#define SEQ 2048
#define MROWS (BATCH*SEQ)   // 4096

// ---- CDNA5 async copy helpers (inline asm; ASYNCcnt-tracked) --------------
static __device__ __forceinline__ void async_copy_b128(unsigned lds_byte_addr,
                                                       const void* gaddr) {
  // GV mode: 64-bit global address in VGPR pair, LDS byte address in VGPR.
  asm volatile("global_load_async_to_lds_b128 %0, %1, off"
               :: "v"(lds_byte_addr), "v"(gaddr) : "memory");
}
static __device__ __forceinline__ void wait_async0() {
  asm volatile("s_wait_asynccnt 0" ::: "memory");
}
static __device__ __forceinline__ unsigned lds_addr(const void* p) {
  // generic LDS pointer = {SHARED_BASE, lds_offset[31:0]} -> truncate
  return (unsigned)(unsigned long long)p;
}

static __device__ __forceinline__ v8f wmma_f16(const F16x16& a, const F16x16& b,
                                               v8f c) {
  return __builtin_amdgcn_wmma_f32_16x16x32_f16(false, a.h, false, b.h,
                                                (short)0, c, false, false);
}

// ---------------------------------------------------------------------------
// Conversions
// ---------------------------------------------------------------------------
__global__ void mha_f32_to_f16(const float* __restrict__ src,
                               h16* __restrict__ dst, int n) {
  int i = blockIdx.x * blockDim.x + threadIdx.x;
  if (i < n) dst[i] = (h16)src[i];
}

// W [512,512] row-major fp32 -> WT [n][k] f16 (transposed)
__global__ void mha_w_transpose(const float* __restrict__ W,
                                h16* __restrict__ WT) {
  int i = blockIdx.x * blockDim.x + threadIdx.x;   // 512*512
  int kk = i >> 9, nn = i & 511;
  WT[nn * 512 + kk] = (h16)W[i];
}

// ---------------------------------------------------------------------------
// WMMA GEMM: C[M,512] = A16[M,512] @ W (BT16 = W^T, f16) + bias
//   mode 0: out16[h][b][l][64] = (c)*scale   (per-head f16 layout)
//   mode 1: out32[row*512+j]   = c + resid[row*512+j]  (fp32, residual)
// Block 128x128, 8 waves (4x2) of 32x64, K-step 32, double-buffered async.
// ---------------------------------------------------------------------------
__global__ __launch_bounds__(256)
void mha_gemm(const h16* __restrict__ A, const h16* __restrict__ BT,
              const float* __restrict__ bias,
              h16* __restrict__ out16, float* __restrict__ out32,
              const float* __restrict__ resid, float scale, int mode) {
  __shared__ h16 As[2 * 128 * 32];
  __shared__ h16 Bs[2 * 128 * 32];

  const int tid = threadIdx.x;
  const int lane = tid & 31;
  const int w = tid >> 5;            // 0..7
  const int wm = w & 3, wn = w >> 2; // 4 x 2 wave grid
  const int m0 = blockIdx.x * 128;
  const int n0 = blockIdx.y * 128;
  const int hl = lane & 15;
  const int hi = lane >> 4;          // 0/1 half-wave

  F32x8 acc[2][4];
#pragma unroll
  for (int ms = 0; ms < 2; ++ms)
#pragma unroll
    for (int ns = 0; ns < 4; ++ns)
#pragma unroll
      for (int r = 0; r < 8; ++r) acc[ms][ns].f[r] = 0.f;

  const unsigned asBase = lds_addr(&As[0]);
  const unsigned bsBase = lds_addr(&Bs[0]);

  auto issue_tile = [&](int buf, int k0) {
#pragma unroll
    for (int j = 0; j < 2; ++j) {
      int c = tid * 2 + j;           // 512 chunks of 8 halves (16B)
      int row = c >> 2;
      int part = (c & 3) * 8;
      unsigned loff = (unsigned)((buf * 4096 + row * 32 + part) * 2);
      async_copy_b128(asBase + loff, A + (size_t)(m0 + row) * 512 + k0 + part);
      async_copy_b128(bsBase + loff, BT + (size_t)(n0 + row) * 512 + k0 + part);
    }
  };

  issue_tile(0, 0);  // prologue

  for (int kt = 0; kt < 16; ++kt) {
    wait_async0();       // current tile landed (only outstanding asyncs)
    __syncthreads();     // visible to all waves; prev buffer fully consumed
    if (kt < 15) issue_tile((kt + 1) & 1, (kt + 1) * 32);  // overlap w/ WMMA

    const h16* Asc = As + (kt & 1) * 4096;
    const h16* Bsc = Bs + (kt & 1) * 4096;

    F16x16 af[2];
#pragma unroll
    for (int ms = 0; ms < 2; ++ms) {
      int mr = wm * 32 + ms * 16 + hl;
      int kb = hi ? 16 : 0;          // A frag: K kb..kb+15 contiguous
      af[ms].u[0] = *(const uint4*)&Asc[mr * 32 + kb];
      af[ms].u[1] = *(const uint4*)&Asc[mr * 32 + kb + 8];
    }
#pragma unroll
    for (int ns = 0; ns < 4; ++ns) {
      int nc = wn * 64 + ns * 16 + hl;
      int kb = hi ? 16 : 0;          // B frag: K kb..kb+15 contiguous
      F16x16 bf;
      bf.u[0] = *(const uint4*)&Bsc[nc * 32 + kb];
      bf.u[1] = *(const uint4*)&Bsc[nc * 32 + kb + 8];
#pragma unroll
      for (int ms = 0; ms < 2; ++ms)
        acc[ms][ns].v = wmma_f16(af[ms], bf, acc[ms][ns].v);
    }
  }

  // epilogue
#pragma unroll
  for (int ms = 0; ms < 2; ++ms)
#pragma unroll
    for (int ns = 0; ns < 4; ++ns) {
      int j = n0 + wn * 64 + ns * 16 + hl;
      float bj = bias[j];
#pragma unroll
      for (int r = 0; r < 8; ++r) {
        int mr = m0 + wm * 32 + ms * 16 + r + hi * 8;
        float val = (acc[ms][ns].f[r] + bj) * scale;
        if (mode == 0) {
          int bb = mr >> 11, ll = mr & 2047;   // row = b*2048 + l
          int hh = j >> 6, dd = j & 63;        // col = h*64 + d
          out16[(((size_t)(hh * BATCH + bb) * SEQ) + ll) * DK + dd] = (h16)val;
        } else {
          size_t o = (size_t)mr * 512 + j;
          out32[o] = val + resid[o];
        }
      }
    }
}

// ---------------------------------------------------------------------------
// Attention: per (hb, 128-row q tile). Wave owns 16 q rows.
// Sweep 1: online row max / sum (flash stats). Sweep 2: write softmax probs
// (non-temporal), accumulate ctx = P @ V with WMMA. Both sweeps use a
// double-buffered async K pipeline; sweep 2 also pipelines the V transpose.
// ---------------------------------------------------------------------------
__global__ __launch_bounds__(256)
void mha_attn(const h16* __restrict__ Qh, const h16* __restrict__ Kh,
              const h16* __restrict__ Vh, const int* __restrict__ mask,
              float* __restrict__ attn, h16* __restrict__ ctx) {
  __shared__ h16 Ks[2 * 64 * 64];   // K tiles, row-major [kk][d], ping-pong
  __shared__ h16 Vt[2 * 64 * 64];   // V tiles transposed [d][kk], ping-pong
  __shared__ h16 Ps[8][16 * 64];    // per-wave P tile [q][kk]

  const int tid = threadIdx.x;
  const int lane = tid & 31;
  const int w = tid >> 5;
  const int hl = lane & 15;
  const int hi = lane >> 4;
  const int hb = blockIdx.y;        // h*BATCH + b
  const int b = hb & 1;
  const int h = hb >> 1;
  const int q0 = blockIdx.x * 128;
  const int qw = q0 + w * 16;

  const h16* Qp = Qh + (size_t)hb * SEQ * DK;
  const h16* Kp = Kh + (size_t)hb * SEQ * DK;
  const h16* Vp = Vh + (size_t)hb * SEQ * DK;
  float* ap = attn + (size_t)hb * SEQ * SEQ;
  const int* mp = mask + (size_t)b * SEQ * SEQ;

  // Q fragments for this wave (held in registers for both sweeps)
  F16x16 qa[2];
#pragma unroll
  for (int ds = 0; ds < 2; ++ds) {
    const h16* p = Qp + (size_t)(qw + hl) * DK + ds * 32;
    int kb = hi ? 16 : 0;
    qa[ds].u[0] = *(const uint4*)(p + kb);
    qa[ds].u[1] = *(const uint4*)(p + kb + 8);
  }

  const unsigned ksBase = lds_addr(&Ks[0]);

  auto issue_k = [&](int buf, int kc) {
#pragma unroll
    for (int j = 0; j < 2; ++j) {
      int c = tid * 2 + j;                       // 512 x 16B = 8KB tile
      async_copy_b128(ksBase + (unsigned)(buf * 8192 + c * 16),
                      Kp + (size_t)kc * 4096 + c * 8);
    }
  };

  float m[8], l[8];
#pragma unroll
  for (int r = 0; r < 8; ++r) { m[r] = -3.0e38f; l[r] = 0.f; }

  // ---------------- Sweep 1: row stats ----------------
  issue_k(0, 0);
  for (int kc = 0; kc < 32; ++kc) {
    wait_async0();
    __syncthreads();
    if (kc < 31) issue_k((kc + 1) & 1, kc + 1);
    const h16* Kc = Ks + (kc & 1) * 4096;

    F32x8 s[4];
#pragma unroll
    for (int n = 0; n < 4; ++n) {
#pragma unroll
      for (int r = 0; r < 8; ++r) s[n].f[r] = 0.f;
      int krow = n * 16 + hl;
#pragma unroll
      for (int ds = 0; ds < 2; ++ds) {
        int db = ds * 32 + (hi ? 16 : 0);
        F16x16 bf;
        bf.u[0] = *(const uint4*)&Kc[krow * 64 + db];
        bf.u[1] = *(const uint4*)&Kc[krow * 64 + db + 8];
        s[n].v = wmma_f16(qa[ds], bf, s[n].v);
      }
      int kcol = kc * 64 + n * 16 + hl;
#pragma unroll
      for (int r = 0; r < 8; ++r) {
        int qr = qw + r + hi * 8;
        if (mp[(size_t)qr * SEQ + kcol] == 0) s[n].f[r] = -1e9f;
      }
    }
    float cm[8];
#pragma unroll
    for (int r = 0; r < 8; ++r)
      cm[r] = fmaxf(fmaxf(s[0].f[r], s[1].f[r]), fmaxf(s[2].f[r], s[3].f[r]));
#pragma unroll
    for (int x = 1; x < 16; x <<= 1)
#pragma unroll
      for (int r = 0; r < 8; ++r)
        cm[r] = fmaxf(cm[r], __shfl_xor(cm[r], x, 32));
#pragma unroll
    for (int r = 0; r < 8; ++r) {
      float mn = fmaxf(m[r], cm[r]);
      float sc = __expf(m[r] - mn);
      float ps = 0.f;
#pragma unroll
      for (int n = 0; n < 4; ++n) ps += __expf(s[n].f[r] - mn);
      l[r] = l[r] * sc + ps;
      m[r] = mn;
    }
  }
  // reduce row sums across the 16 lanes sharing each row
#pragma unroll
  for (int x = 1; x < 16; x <<= 1)
#pragma unroll
    for (int r = 0; r < 8; ++r) l[r] += __shfl_xor(l[r], x, 32);
  float invl[8];
#pragma unroll
  for (int r = 0; r < 8; ++r) invl[r] = 1.0f / l[r];

  F32x8 cacc[4];
#pragma unroll
  for (int n = 0; n < 4; ++n)
#pragma unroll
    for (int r = 0; r < 8; ++r) cacc[n].f[r] = 0.f;

  // ---------------- Sweep 2: probs + P@V ----------------
  // prologue: V tile 0 -> Vt[0] (transposed), K tile 0 async
  {
    const uint4* vs = (const uint4*)(Vp + 0);
    F16x16 tmp;
    tmp.u[0] = vs[tid * 2];
    tmp.u[1] = vs[tid * 2 + 1];
    int kk = tid >> 2;
    int d0 = (tid & 3) * 16;
#pragma unroll
    for (int i = 0; i < 16; ++i) Vt[(d0 + i) * 64 + kk] = tmp.e[i];
  }
  issue_k(0, 0);

  for (int kc = 0; kc < 32; ++kc) {
    wait_async0();
    __syncthreads();     // K/V tiles (cur) visible; (nxt) buffers free
    const int haveNext = (kc < 31);
    F16x16 vreg;
    if (haveNext) {
      issue_k((kc + 1) & 1, kc + 1);
      // V loads for next tile into registers; latency hidden by S/P compute
      const uint4* vs = (const uint4*)(Vp + (size_t)(kc + 1) * 4096);
      vreg.u[0] = vs[tid * 2];
      vreg.u[1] = vs[tid * 2 + 1];
    }
    const h16* Kc = Ks + (kc & 1) * 4096;
    const h16* Vc = Vt + (kc & 1) * 4096;

    F32x8 s[4];
#pragma unroll
    for (int n = 0; n < 4; ++n) {
#pragma unroll
      for (int r = 0; r < 8; ++r) s[n].f[r] = 0.f;
      int krow = n * 16 + hl;
#pragma unroll
      for (int ds = 0; ds < 2; ++ds) {
        int db = ds * 32 + (hi ? 16 : 0);
        F16x16 bf;
        bf.u[0] = *(const uint4*)&Kc[krow * 64 + db];
        bf.u[1] = *(const uint4*)&Kc[krow * 64 + db + 8];
        s[n].v = wmma_f16(qa[ds], bf, s[n].v);
      }
      int kcol = kc * 64 + n * 16 + hl;
#pragma unroll
      for (int r = 0; r < 8; ++r) {
        int qsub = r + hi * 8;
        int qr = qw + qsub;
        float sv = s[n].f[r];
        if (mp[(size_t)qr * SEQ + kcol] == 0) sv = -1e9f;
        float p = __expf(sv - m[r]) * invl[r];
        // streamed, never re-read: non-temporal store keeps L2 for mask/K/V
        __builtin_nontemporal_store(p, &ap[(size_t)qr * SEQ + kcol]);
        Ps[w][qsub * 64 + n * 16 + hl] = (h16)p;         // stash for P@V
      }
    }
    // stage next V tile into the free buffer (load latency already hidden)
    if (haveNext) {
      h16* Vn = Vt + ((kc + 1) & 1) * 4096;
      int kk = tid >> 2;
      int d0 = (tid & 3) * 16;
#pragma unroll
      for (int i = 0; i < 16; ++i) Vn[(d0 + i) * 64 + kk] = vreg.e[i];
    }
    // P @ V  (per-wave LDS tile; ds ordering within wave handled by compiler)
#pragma unroll
    for (int n2 = 0; n2 < 4; ++n2) {
      int vcol = n2 * 16 + hl;
#pragma unroll
      for (int ks = 0; ks < 2; ++ks) {
        F16x16 pf, vf;
        int kbp = ks * 32 + (hi ? 16 : 0);
        pf.u[0] = *(const uint4*)&Ps[w][hl * 64 + kbp];
        pf.u[1] = *(const uint4*)&Ps[w][hl * 64 + kbp + 8];
        int kbv = ks * 32 + (hi ? 16 : 0);
        vf.u[0] = *(const uint4*)&Vc[vcol * 64 + kbv];
        vf.u[1] = *(const uint4*)&Vc[vcol * 64 + kbv + 8];
        cacc[n2].v = wmma_f16(pf, vf, cacc[n2].v);
      }
    }
  }

  // write context rows [b*2048+q][h*64+v] (f16)
#pragma unroll
  for (int n2 = 0; n2 < 4; ++n2) {
    int col = h * 64 + n2 * 16 + hl;
#pragma unroll
    for (int r = 0; r < 8; ++r) {
      int qr = qw + r + hi * 8;
      ctx[((size_t)(b * SEQ + qr)) * 512 + col] = (h16)cacc[n2].f[r];
    }
  }
}

// ---------------------------------------------------------------------------
extern "C" void kernel_launch(void* const* d_in, const int* in_sizes, int n_in,
                              void* d_out, int out_size, void* d_ws,
                              size_t ws_size, hipStream_t stream) {
  const float* q  = (const float*)d_in[0];
  const float* k  = (const float*)d_in[1];
  const float* v  = (const float*)d_in[2];
  const int* mask = (const int*)d_in[3];
  const float* Wq = (const float*)d_in[4];
  const float* bq = (const float*)d_in[5];
  const float* Wk = (const float*)d_in[6];
  const float* bk = (const float*)d_in[7];
  const float* Wv = (const float*)d_in[8];
  const float* bv = (const float*)d_in[9];
  const float* Wo = (const float*)d_in[10];
  const float* bo = (const float*)d_in[11];

  const size_t MB = 1u << 20;
  char* ws = (char*)d_ws;
  h16* qf  = (h16*)(ws + 0 * MB);
  h16* kf  = (h16*)(ws + 4 * MB);
  h16* vf  = (h16*)(ws + 8 * MB);
  h16* WqT = (h16*)(ws + 12 * MB);
  h16* WkT = (h16*)(ws + 12 * MB + 512 * 1024);
  h16* WvT = (h16*)(ws + 13 * MB);
  h16* WoT = (h16*)(ws + 13 * MB + 512 * 1024);
  h16* Qh  = (h16*)(ws + 14 * MB);
  h16* Kh  = (h16*)(ws + 18 * MB);
  h16* Vh  = (h16*)(ws + 22 * MB);
  h16* ctx = (h16*)(ws + 26 * MB);

  float* outp = (float*)d_out;
  float* attnp = outp + (size_t)BATCH * SEQ * DMODEL;

  const int NX = MROWS * DMODEL;  // 2,097,152
  mha_f32_to_f16<<<NX / 256, 256, 0, stream>>>(q, qf, NX);
  mha_f32_to_f16<<<NX / 256, 256, 0, stream>>>(k, kf, NX);
  mha_f32_to_f16<<<NX / 256, 256, 0, stream>>>(v, vf, NX);
  mha_w_transpose<<<(512 * 512) / 256, 256, 0, stream>>>(Wq, WqT);
  mha_w_transpose<<<(512 * 512) / 256, 256, 0, stream>>>(Wk, WkT);
  mha_w_transpose<<<(512 * 512) / 256, 256, 0, stream>>>(Wv, WvT);
  mha_w_transpose<<<(512 * 512) / 256, 256, 0, stream>>>(Wo, WoT);

  dim3 gg(MROWS / 128, DMODEL / 128);
  // fold reference temperature (1/d_k) into Q projection
  mha_gemm<<<gg, 256, 0, stream>>>(qf, WqT, bq, Qh, nullptr, nullptr,
                                   1.0f / 64.0f, 0);
  mha_gemm<<<gg, 256, 0, stream>>>(kf, WkT, bk, Kh, nullptr, nullptr, 1.0f, 0);
  mha_gemm<<<gg, 256, 0, stream>>>(vf, WvT, bv, Vh, nullptr, nullptr, 1.0f, 0);

  dim3 ga(SEQ / 128, NHEAD * BATCH);
  mha_attn<<<ga, 256, 0, stream>>>(Qh, Kh, Vh, mask, attnp, ctx);

  mha_gemm<<<gg, 256, 0, stream>>>(ctx, WoT, bo, nullptr, outp, q, 1.0f, 1);
}
